// GraphAnomalyDetectionModel_83056077570956
// MI455X (gfx1250) — matrix operand order repaired and checked by reading, not compiled, not run on hardware
//
#include <hip/hip_runtime.h>
#include <hip/hip_bf16.h>
#include <math.h>

// ---------------------------------------------------------------------------
// Types for CDNA5 WMMA (wave32): D(16x16,f32) = A(16x32,bf16) x B(32x16,bf16) + C
// ---------------------------------------------------------------------------
typedef __attribute__((ext_vector_type(16))) __bf16 v16bf;
typedef __attribute__((ext_vector_type(8)))  float  v8f;
typedef __attribute__((ext_vector_type(4)))  unsigned int u32x4;
typedef __attribute__((ext_vector_type(8)))  int    i32x8;
typedef __attribute__((ext_vector_type(4)))  int    i32x4;

#define WMMA_BF16(a, b, c) \
  __builtin_amdgcn_wmma_f32_16x16x32_bf16(false, (a), false, (b), (short)0, (c), false, false)

// Problem constants (match reference)
#define N_USERS 500000
#define N_ITEMS 50000
#define N_EDGES 1000000
#define HC 128   // heads * channels

// ---------------------------------------------------------------------------
// A-fragment loader (ISA 7.12.2 layout, wave32), row-major fp32 source:
//   lane<16 : elems 0..7 = K k0+0..7,  elems 8..15 = K k0+16..23  (row = lane&15)
//   lane>=16: elems 0..7 = K k0+8..15, elems 8..15 = K k0+24..31
// ---------------------------------------------------------------------------
template <int K, int LDA>
__device__ __forceinline__ v16bf load_a_frag(const float* __restrict__ A,
                                             int m0, int k0, int lane) {
  int r = m0 + (lane & 15);
  int base = k0 + ((lane >> 4) << 3);
  const float* row = A + (size_t)r * LDA;
  v16bf a;
  if ((k0 + 32 <= K) && (LDA % 4 == 0)) {
    // fully in-range, 16B-aligned: four b128 loads
    const float4* p = (const float4*)(row + base);
    float4 q0 = p[0], q1 = p[1], q2 = p[4], q3 = p[5];
    a[0]=(__bf16)q0.x; a[1]=(__bf16)q0.y; a[2]=(__bf16)q0.z; a[3]=(__bf16)q0.w;
    a[4]=(__bf16)q1.x; a[5]=(__bf16)q1.y; a[6]=(__bf16)q1.z; a[7]=(__bf16)q1.w;
    a[8]=(__bf16)q2.x; a[9]=(__bf16)q2.y; a[10]=(__bf16)q2.z; a[11]=(__bf16)q2.w;
    a[12]=(__bf16)q3.x; a[13]=(__bf16)q3.y; a[14]=(__bf16)q3.z; a[15]=(__bf16)q3.w;
  } else if (k0 + 32 <= K) {
    // fully in-range, unknown alignment: unconditional scalar loads (batched)
#pragma unroll
    for (int i = 0; i < 8; ++i) {
      a[i]     = (__bf16)row[base + i];
      a[i + 8] = (__bf16)row[base + 16 + i];
    }
  } else {
    // tail chunk: clamp address (always safe), select value -> v_cndmask
#pragma unroll
    for (int i = 0; i < 8; ++i) {
      int k1 = base + i, k2 = base + 16 + i;
      float v1 = row[k1 < K ? k1 : K - 1];
      float v2 = row[k2 < K ? k2 : K - 1];
      a[i]     = (__bf16)(k1 < K ? v1 : 0.0f);
      a[i + 8] = (__bf16)(k2 < K ? v2 : 0.0f);
    }
  }
  return a;
}

// ---------------------------------------------------------------------------
// Generic WMMA GEMM: D[M x NT*16] = act(A[M x K] @ B[K x NT*16] + bias)
// B is DMA'd into LDS once per block by the Tensor Data Mover (wave 0 issues
// TENSOR_LOAD_TO_LDS, waits on TENSORcnt), converted to WMMA fragment layout
// in LDS, then every wave reads fragments back with ds_load_b128.  TDM OOB
// reads return zero, so rows K..KPAD-1 are zero-filled (free K-tail padding).
// One wave handles a 16-row strip across the full N.  M multiple of 16.
// ---------------------------------------------------------------------------
template <int NT, int K, int LDA>
__global__ void wmma_gemm_bias(const float* __restrict__ A,
                               const float* __restrict__ B, int ldb,
                               const float* __restrict__ bias,
                               float* __restrict__ D, int ldd,
                               int M, int relu_flag) {
  constexpr int KSTEPS = (K + 31) / 32;
  constexpr int KPAD = KSTEPS * 32;
  constexpr int N = NT * 16;
  constexpr int NFRAG = KSTEPS * NT;
  __shared__ float Braw[KPAD][N];      // raw fp32 weight tile (TDM destination)
  __shared__ v16bf bfrag[NFRAG][32];   // 32 B per lane per fragment

  int lane = threadIdx.x & 31;
  int wave = threadIdx.x >> 5;

  if (wave == 0) {
    // ---- Tensor DMA: global (K x ldb fp32, row-major) -> LDS tile ----
    unsigned lds_addr = (unsigned)(unsigned long long)(uintptr_t)&Braw[0][0];
    unsigned long long ga = (unsigned long long)(uintptr_t)B;
    unsigned dim0 = (unsigned)ldb;          // tensor row length (elements)
    unsigned dim1 = (unsigned)K;            // tensor rows
    unsigned long long stride0 = (unsigned long long)ldb;
    u32x4 g0;
    g0[0] = 1u;                              // count=1, user mode, no gather
    g0[1] = lds_addr;                        // D#.lds_addr
    g0[2] = (unsigned)(ga & 0xFFFFFFFFu);    // global_addr[31:0]
    g0[3] = (unsigned)((ga >> 32) & 0x01FFFFFFu) | (2u << 30);  // [56:32] | type=2
    i32x8 g1;
    g1[0] = (int)(2u << 16);                 // data_size=2 (4 bytes)
    g1[1] = (int)((dim0 & 0xFFFFu) << 16);   // tensor_dim0[15:0] (bits 63:48)
    g1[2] = (int)(((dim0 >> 16) & 0xFFFFu) | ((dim1 & 0xFFFFu) << 16));
    g1[3] = (int)(((dim1 >> 16) & 0xFFFFu) | ((unsigned)N << 16));   // tile_dim0
    g1[4] = (int)((unsigned)KPAD);           // tile_dim1 (padded rows), tile_dim2=0
    g1[5] = (int)(unsigned)(stride0 & 0xFFFFFFFFull);                // dim0_stride lo
    g1[6] = (int)(unsigned)((stride0 >> 32) & 0xFFFFull);            // dim0_stride hi
    g1[7] = 0;
    i32x4 gz; gz[0] = 0; gz[1] = 0; gz[2] = 0; gz[3] = 0;
    i32x8 gz8; gz8[0] = 0; gz8[1] = 0; gz8[2] = 0; gz8[3] = 0;
    gz8[4] = 0; gz8[5] = 0; gz8[6] = 0; gz8[7] = 0;
    __builtin_amdgcn_tensor_load_to_lds(g0, g1, gz, gz, gz8, 0);
    __builtin_amdgcn_s_wait_tensorcnt(0);

    // ---- convert raw tile to WMMA fragment layout (no tail checks needed) ----
#pragma unroll 1
    for (int f = 0; f < NFRAG; ++f) {
      int ks = f / NT, n = f - ks * NT;
      int col = n * 16 + (lane & 15);
      int kb = ks * 32 + ((lane >> 4) << 3);
      v16bf b;
#pragma unroll
      for (int e = 0; e < 16; ++e) b[e] = (__bf16)Braw[kb + e + ((e >> 3) << 3)][col];
      bfrag[f][lane] = b;
    }
  }
  __syncthreads();

  // ---- per-wave 16-row strip (clamped so all waves keep EXEC all-ones) ----
  int tile = blockIdx.x * (blockDim.x >> 5) + wave;
  bool valid = tile * 16 < M;
  int m0 = valid ? tile * 16 : M - 16;

  v8f acc[NT];
#pragma unroll
  for (int n = 0; n < NT; ++n) {
    float bv = bias ? bias[n * 16 + (lane & 15)] : 0.0f;
#pragma unroll
    for (int v = 0; v < 8; ++v) acc[n][v] = bv;
  }

#pragma unroll
  for (int ks = 0; ks < KSTEPS; ++ks) {
    v16bf a = load_a_frag<K, LDA>(A, m0, ks * 32, lane);
#pragma unroll
    for (int n = 0; n < NT; ++n) {
      v16bf b = bfrag[ks * NT + n][lane];            // 2x ds_load_b128
      acc[n] = WMMA_BF16(a, b, acc[n]);
    }
  }

  if (valid) {
    int rbase = (lane >> 4) * 8;
    int col = lane & 15;
#pragma unroll
    for (int n = 0; n < NT; ++n)
#pragma unroll
      for (int v = 0; v < 8; ++v) {
        float x = acc[n][v];
        if (relu_flag) x = fmaxf(x, 0.0f);
        D[(size_t)(m0 + rbase + v) * ldd + n * 16 + col] = x;
      }
  }
}

// ---------------------------------------------------------------------------
// Utility kernels
// ---------------------------------------------------------------------------
__global__ void fill_kernel(float4* __restrict__ p, float v, size_t n4) {
  size_t i = (size_t)blockIdx.x * blockDim.x + threadIdx.x;
  if (i < n4) p[i] = make_float4(v, v, v, v);
}

__global__ void bias_act_kernel(const float4* __restrict__ in,
                                const float4* __restrict__ bias,  // 32 float4s
                                float4* __restrict__ out, size_t n4, int relu) {
  size_t i = (size_t)blockIdx.x * blockDim.x + threadIdx.x;
  if (i >= n4) return;
  float4 x = in[i];
  float4 b = bias[i & (HC / 4 - 1)];
  x.x += b.x; x.y += b.y; x.z += b.z; x.w += b.w;
  if (relu) {
    x.x = fmaxf(x.x, 0.f); x.y = fmaxf(x.y, 0.f);
    x.z = fmaxf(x.z, 0.f); x.w = fmaxf(x.w, 0.f);
  }
  out[i] = x;
}

__device__ __forceinline__ void atomicMaxFloat(float* addr, float val) {
  if (val >= 0.0f)
    atomicMax((int*)addr, __float_as_int(val));
  else
    atomicMin((unsigned int*)addr, (unsigned int)__float_as_int(val));
}

// ---------------------------------------------------------------------------
// Edge kernels (GATv2).  One wave per edge; lane owns 4 contiguous channels.
// Heads: channel c -> head c>>5; lanes 8h..8h+7 cover head h.
// ---------------------------------------------------------------------------
__global__ void edge_score_kernel(const float* __restrict__ xl,
                                  const float* __restrict__ xr,
                                  const int* __restrict__ gsrc,
                                  const int* __restrict__ gdst,
                                  const float* __restrict__ eattr,   // [E,3] or null
                                  const float* __restrict__ We,      // [3,128] or null
                                  const float* __restrict__ att,     // [4,32]
                                  float* __restrict__ escore,        // [E,4]
                                  float* __restrict__ mbuf,          // [n_dst,4], pre -inf
                                  int E) {
  int lane = threadIdx.x & 31;
  int e = blockIdx.x * (blockDim.x >> 5) + (threadIdx.x >> 5);
  if (e >= E) return;
  int s = gsrc[e], d = gdst[e];
  // prefetch a future edge's source row into GL2 (gather addresses known early)
  int ep = e + 64;
  if (ep < E) __builtin_prefetch(xl + (size_t)gsrc[ep] * HC, 0, 1);

  int c0 = lane * 4;
  int head = c0 >> 5;
  float4 xl4 = *(const float4*)(xl + (size_t)s * HC + c0);
  float4 xr4 = *(const float4*)(xr + (size_t)d * HC + c0);
  float4 at4 = *(const float4*)(att + (head << 5) + (c0 & 31));
  float4 h4 = make_float4(xl4.x + xr4.x, xl4.y + xr4.y, xl4.z + xr4.z, xl4.w + xr4.w);
  if (We) {
    float ea0 = eattr[(size_t)e * 3], ea1 = eattr[(size_t)e * 3 + 1], ea2 = eattr[(size_t)e * 3 + 2];
    float4 w0 = *(const float4*)(We + c0);
    float4 w1 = *(const float4*)(We + HC + c0);
    float4 w2 = *(const float4*)(We + 2 * HC + c0);
    h4.x += ea0 * w0.x + ea1 * w1.x + ea2 * w2.x;
    h4.y += ea0 * w0.y + ea1 * w1.y + ea2 * w2.y;
    h4.z += ea0 * w0.z + ea1 * w1.z + ea2 * w2.z;
    h4.w += ea0 * w0.w + ea1 * w1.w + ea2 * w2.w;
  }
  // leaky_relu(0.2) then weight by att, accumulate
  float part = (h4.x > 0.f ? h4.x : 0.2f * h4.x) * at4.x
             + (h4.y > 0.f ? h4.y : 0.2f * h4.y) * at4.y
             + (h4.z > 0.f ? h4.z : 0.2f * h4.z) * at4.z
             + (h4.w > 0.f ? h4.w : 0.2f * h4.w) * at4.w;
  // reduce over the 8 lanes of this head
  part += __shfl_xor(part, 1, 32);
  part += __shfl_xor(part, 2, 32);
  part += __shfl_xor(part, 4, 32);
  if ((lane & 7) == 0) {
    escore[(size_t)e * 4 + head] = part;
    atomicMaxFloat(&mbuf[(size_t)d * 4 + head], part);
  }
}

// one thread per edge: all 4 heads via float4
__global__ void edge_exp_kernel(float* __restrict__ escore,
                                const float* __restrict__ mbuf,
                                float* __restrict__ sbuf,
                                const int* __restrict__ gdst, int E) {
  int e = blockIdx.x * blockDim.x + threadIdx.x;
  if (e >= E) return;
  int d = gdst[e];
  float4 sc = *(const float4*)(escore + (size_t)e * 4);
  float4 mm = *(const float4*)(mbuf + (size_t)d * 4);
  float4 ex = make_float4(__expf(sc.x - mm.x), __expf(sc.y - mm.y),
                          __expf(sc.z - mm.z), __expf(sc.w - mm.w));
  *(float4*)(escore + (size_t)e * 4) = ex;
  float* sd = sbuf + (size_t)d * 4;
  atomicAdd(sd + 0, ex.x);
  atomicAdd(sd + 1, ex.y);
  atomicAdd(sd + 2, ex.z);
  atomicAdd(sd + 3, ex.w);
}

__global__ void edge_aggr_kernel(const float* __restrict__ escore,
                                 const float* __restrict__ sbuf,
                                 const float* __restrict__ xl,
                                 const int* __restrict__ gsrc,
                                 const int* __restrict__ gdst,
                                 float* __restrict__ acc, int E) {
  int lane = threadIdx.x & 31;
  int e = blockIdx.x * (blockDim.x >> 5) + (threadIdx.x >> 5);
  if (e >= E) return;
  int s = gsrc[e], d = gdst[e];
  int c0 = lane * 4;
  int head = c0 >> 5;
  float alpha = escore[(size_t)e * 4 + head] /
                (sbuf[(size_t)d * 4 + head] + 1e-16f);
  float4 x4 = *(const float4*)(xl + (size_t)s * HC + c0);
  float* accd = acc + (size_t)d * HC + c0;
  atomicAdd(accd + 0, alpha * x4.x);
  atomicAdd(accd + 1, alpha * x4.y);
  atomicAdd(accd + 2, alpha * x4.z);
  atomicAdd(accd + 3, alpha * x4.w);
}

// classifier tail: scores = sigmoid(hid[64] @ W2 + b2)
__global__ void cls_out_kernel(const float* __restrict__ hid,
                               const float* __restrict__ W2,
                               const float* __restrict__ b2,
                               float* __restrict__ out, int M) {
  int r = blockIdx.x * blockDim.x + threadIdx.x;
  if (r >= M) return;
  float a = b2[0];
  const float4* hr = (const float4*)(hid + (size_t)r * 64);
  const float4* w4 = (const float4*)W2;
#pragma unroll
  for (int j = 0; j < 16; ++j) {
    float4 h = hr[j], w = w4[j];
    a += h.x * w.x + h.y * w.y + h.z * w.z + h.w * w.w;
  }
  out[r] = 1.0f / (1.0f + __expf(-a));
}

// ---------------------------------------------------------------------------
// Launch
// ---------------------------------------------------------------------------
static inline int waveGrid(int nTiles) { return (nTiles + 7) / 8; }  // 8 waves/block

extern "C" void kernel_launch(void* const* d_in, const int* in_sizes, int n_in,
                              void* d_out, int out_size, void* d_ws, size_t ws_size,
                              hipStream_t stream) {
  const float* customer_x = (const float*)d_in[0];
  const float* fund_x     = (const float*)d_in[1];
  const float* edge_attr  = (const float*)d_in[2];
  const int*   edge_src   = (const int*)d_in[3];
  const int*   edge_dst   = (const int*)d_in[4];
  const float* user_W = (const float*)d_in[5];  const float* user_b = (const float*)d_in[6];
  const float* item_W = (const float*)d_in[7];  const float* item_b = (const float*)d_in[8];
  const float* c1_Wl  = (const float*)d_in[9];  const float* c1_bl  = (const float*)d_in[10];
  const float* c1_Wr  = (const float*)d_in[11]; const float* c1_br  = (const float*)d_in[12];
  const float* c1_We  = (const float*)d_in[13];
  const float* c1_att = (const float*)d_in[14]; const float* c1_bias = (const float*)d_in[15];
  const float* c2_Wl  = (const float*)d_in[16]; const float* c2_bl  = (const float*)d_in[17];
  const float* c2_Wr  = (const float*)d_in[18]; const float* c2_br  = (const float*)d_in[19];
  const float* c2_att = (const float*)d_in[20]; const float* c2_bias = (const float*)d_in[21];
  const float* cls_W1 = (const float*)d_in[22]; const float* cls_b1 = (const float*)d_in[23];
  const float* cls_W2 = (const float*)d_in[24]; const float* cls_b2 = (const float*)d_in[25];
  float* scores = (float*)d_out;

  // Workspace layout (floats); all region sizes are multiples of 4 -> 16B aligned
  float* ws = (float*)d_ws;
  size_t o = 0;
  float* user_x   = ws + o; o += (size_t)N_USERS * 32;
  float* xl1      = ws + o; o += (size_t)N_USERS * HC;   // reused later as user_h
  float* xr2      = ws + o; o += (size_t)N_USERS * HC;   // first half reused as hid
  float* item_x   = ws + o; o += (size_t)N_ITEMS * 32;
  float* xr1      = ws + o; o += (size_t)N_ITEMS * HC;
  float* item_acc = ws + o; o += (size_t)N_ITEMS * HC;
  float* item_h   = ws + o; o += (size_t)N_ITEMS * HC;
  float* xl2      = ws + o; o += (size_t)N_ITEMS * HC;
  float* e1       = ws + o; o += (size_t)N_EDGES * 4;
  float* m1       = ws + o; o += (size_t)N_ITEMS * 4;
  float* s1       = ws + o; o += (size_t)N_ITEMS * 4;
  float* e2       = ws + o; o += (size_t)N_EDGES * 4;
  float* m2       = ws + o; o += (size_t)N_USERS * 4;
  float* s2       = ws + o; o += (size_t)N_USERS * 4;
  float* user_acc = ws + o; o += (size_t)N_USERS * HC;
  float* user_h = xl1;  // alias: xl1 dead after conv1 aggregation
  float* hid    = xr2;  // alias: xr2 dead after conv2 scoring

  const float NEG_INF = -__builtin_inff();
  auto fill = [&](float* p, float v, size_t n) {
    size_t n4 = n / 4;
    fill_kernel<<<(unsigned)((n4 + 255) / 256), 256, 0, stream>>>((float4*)p, v, n4);
  };
  // Re-init accumulators every call (graph-replay safe)
  fill(m1, NEG_INF, (size_t)N_ITEMS * 4);
  fill(s1, 0.0f,    (size_t)N_ITEMS * 4);
  fill(item_acc, 0.0f, (size_t)N_ITEMS * HC);
  fill(m2, NEG_INF, (size_t)N_USERS * 4);
  fill(s2, 0.0f,    (size_t)N_USERS * 4);
  fill(user_acc, 0.0f, (size_t)N_USERS * HC);

  const int uTiles = N_USERS / 16, iTiles = N_ITEMS / 16;

  // Input projections
  wmma_gemm_bias<2, 101, 101><<<waveGrid(uTiles), 256, 0, stream>>>(
      customer_x, user_W, 32, user_b, user_x, 32, N_USERS, 0);
  wmma_gemm_bias<2, 1, 1><<<waveGrid(iTiles), 256, 0, stream>>>(
      fund_x, item_W, 32, item_b, item_x, 32, N_ITEMS, 0);

  // conv1 linear parts
  wmma_gemm_bias<8, 32, 32><<<waveGrid(uTiles), 256, 0, stream>>>(
      user_x, c1_Wl, HC, c1_bl, xl1, HC, N_USERS, 0);
  wmma_gemm_bias<8, 32, 32><<<waveGrid(iTiles), 256, 0, stream>>>(
      item_x, c1_Wr, HC, c1_br, xr1, HC, N_ITEMS, 0);
  // conv2 right side (from user_x) can go early
  wmma_gemm_bias<8, 32, 32><<<waveGrid(uTiles), 256, 0, stream>>>(
      user_x, c2_Wr, HC, c2_br, xr2, HC, N_USERS, 0);

  // conv1 edge phase: dst = items
  int eGrid = (N_EDGES + 7) / 8;
  edge_score_kernel<<<eGrid, 256, 0, stream>>>(xl1, xr1, edge_src, edge_dst,
      edge_attr, c1_We, c1_att, e1, m1, N_EDGES);
  edge_exp_kernel<<<(N_EDGES + 255) / 256, 256, 0, stream>>>(e1, m1, s1, edge_dst, N_EDGES);
  edge_aggr_kernel<<<eGrid, 256, 0, stream>>>(e1, s1, xl1, edge_src, edge_dst,
      item_acc, N_EDGES);
  bias_act_kernel<<<(unsigned)(((size_t)N_ITEMS * HC / 4 + 255) / 256), 256, 0, stream>>>(
      (const float4*)item_acc, (const float4*)c1_bias, (float4*)item_h,
      (size_t)N_ITEMS * HC / 4, 1);  // relu

  // conv2 left side
  wmma_gemm_bias<8, 128, 128><<<waveGrid(iTiles), 256, 0, stream>>>(
      item_h, c2_Wl, HC, c2_bl, xl2, HC, N_ITEMS, 0);

  // conv2 edge phase: flipped edges, dst = users, no edge features
  edge_score_kernel<<<eGrid, 256, 0, stream>>>(xl2, xr2, edge_dst, edge_src,
      nullptr, nullptr, c2_att, e2, m2, N_EDGES);
  edge_exp_kernel<<<(N_EDGES + 255) / 256, 256, 0, stream>>>(e2, m2, s2, edge_src, N_EDGES);
  edge_aggr_kernel<<<eGrid, 256, 0, stream>>>(e2, s2, xl2, edge_dst, edge_src,
      user_acc, N_EDGES);
  bias_act_kernel<<<(unsigned)(((size_t)N_USERS * HC / 4 + 255) / 256), 256, 0, stream>>>(
      (const float4*)user_acc, (const float4*)c2_bias, (float4*)user_h,
      (size_t)N_USERS * HC / 4, 0);  // no relu

  // classifier
  wmma_gemm_bias<4, 128, 128><<<waveGrid(uTiles), 256, 0, stream>>>(
      user_h, cls_W1, 64, cls_b1, hid, 64, N_USERS, 1);  // relu
  cls_out_kernel<<<(N_USERS + 255) / 256, 256, 0, stream>>>(hid, cls_W2, cls_b2,
      scores, N_USERS);
}
